// QloraLinear_84851373900522
// MI455X (gfx1250) — compile-verified, matrix operand order checked by reading, not run on hardware
//
#include <hip/hip_runtime.h>
#include <hip/hip_bf16.h>

// ---------------------------------------------------------------------------
// QLoRA linear for MI455X (gfx1250, wave32, WMMA).
// out = x @ dequant_nf4(codes, scales)^T + 2.0 * (x @ lora_A^T) @ lora_B^T
// Compute-bound (275 GFLOP vs ~14us HBM) -> bf16 WMMA, f32 accumulate.
// Double-buffered LDS software pipeline (1 barrier/K-step, global-load
// latency hidden under WMMAs), native __bf16 packed converts, NF4 dequant
// via ds_bpermute register table (no LDS bank conflicts, no index mask —
// wave32 bpermute uses index bits [6:2] only and codes are < 16).
// ---------------------------------------------------------------------------

typedef __attribute__((ext_vector_type(16))) __bf16 v16bf;
typedef __attribute__((ext_vector_type(4)))  __bf16 bf16x4;
typedef __attribute__((ext_vector_type(2)))  __bf16 bf16x2;
typedef __attribute__((ext_vector_type(8)))  float  v8f;

#define LDSS 40          // LDS row stride in bf16 elems (80B: 16B aligned, bank-spread)
#define KDIM 4096
#define NDIM 4096

__constant__ float NF4_TAB[16] = {
    -1.0f, -0.6961928009986877f, -0.5250730514526367f, -0.39491748809814453f,
    -0.28444138169288635f, -0.18477343022823334f, -0.09105003625154495f, 0.0f,
    0.07958029955625534f, 0.16093020141124725f, 0.24611230194568634f,
    0.33791524171829224f, 0.44070982933044434f, 0.5626170039176941f,
    0.7229568362236023f, 1.0f};

struct __align__(16) U4 { unsigned a, b, c, d; };
union FragU { U4 q[2]; v16bf v; };

// 16x32 bf16 A/B fragment from an LDS tile stored as [row][k] rows.
// ISA layout: lanes 0-15 hold row=lane, K halves {0..7,16..23};
// lanes 16-31 hold row=lane-16, K halves {8..15,24..31}. Two ds_load_b128.
__device__ inline v16bf ld_frag(const unsigned short* base, int row0, int lane) {
  const unsigned short* p =
      base + (unsigned)(row0 + (lane & 15)) * LDSS + ((lane >> 4) << 3);
  FragU u;
  u.q[0] = *(const U4*)p;
  u.q[1] = *(const U4*)(p + 16);
  return u.v;
}

// NF4 code -> value via lane-striped register table (ds_bpermute, no LDS mem).
// wave32 bpermute uses only index bits [6:2]; codes are 0..15 so no mask.
__device__ inline float nf4_val(float tabreg, int code) {
  int r = __builtin_amdgcn_ds_bpermute(code << 2,
                                       __builtin_bit_cast(int, tabreg));
  return __builtin_bit_cast(float, r);
}

__device__ inline bf16x4 cvt4(float a, float b, float c, float d) {
  bf16x4 v; v.x = (__bf16)a; v.y = (__bf16)b; v.z = (__bf16)c; v.w = (__bf16)d;
  return v;
}

// In-flight registers for the next K-step's tiles.
struct TileRegs {
  float4 xv[4];   // X tile rows (f32)
  int4   cd[4];   // NF4 codes
  float  sc[4];   // per-row scales
  float2 la;      // lora_A slice
};

__device__ inline void issue_loads(TileRegs& r, const float* __restrict__ x,
                                   const int* __restrict__ codes,
                                   const float* __restrict__ scales,
                                   const float* __restrict__ lora_A,
                                   int tileM, int tileN, int k0, int tid) {
  const int c4 = (tid & 7) * 4, rbase = tid >> 3;
#pragma unroll
  for (int p = 0; p < 4; ++p) {
    int row = rbase + p * 32;
    r.xv[p] = *(const float4*)(x + (size_t)(tileM + row) * KDIM + k0 + c4);
    r.cd[p] = *(const int4*)(codes + (size_t)(tileN + row) * KDIM + k0 + c4);
    r.sc[p] = scales[(size_t)(tileN + row) * (KDIM / 64) + (k0 >> 6)];
  }
  int rr = tid >> 4, kk = (tid & 15) * 2;
  r.la = *(const float2*)(lora_A + (size_t)rr * KDIM + k0 + kk);
}

__device__ inline void store_tiles(const TileRegs& r, unsigned short* bx,
                                   unsigned short* bw, unsigned short* bla,
                                   float tabreg, int tid) {
  const int c4 = (tid & 7) * 4, rbase = tid >> 3;
#pragma unroll
  for (int p = 0; p < 4; ++p) {
    int row = rbase + p * 32;
    *(bf16x4*)&bx[row * LDSS + c4] =
        cvt4(r.xv[p].x, r.xv[p].y, r.xv[p].z, r.xv[p].w);
    float s = r.sc[p];
    *(bf16x4*)&bw[row * LDSS + c4] =
        cvt4(nf4_val(tabreg, r.cd[p].x) * s, nf4_val(tabreg, r.cd[p].y) * s,
             nf4_val(tabreg, r.cd[p].z) * s, nf4_val(tabreg, r.cd[p].w) * s);
  }
  int rr = tid >> 4, kk = (tid & 15) * 2;
  bf16x2 lv; lv.x = (__bf16)r.la.x; lv.y = (__bf16)r.la.y;
  *(bf16x2*)&bla[rr * LDSS + kk] = lv;
}

__global__ __launch_bounds__(256)
void qlora_wmma_kernel(const float* __restrict__ x,
                       const int*   __restrict__ codes,
                       const float* __restrict__ scales,
                       const float* __restrict__ lora_A,
                       const float* __restrict__ lora_B,
                       float* __restrict__ out) {
  // Double-buffered main tiles; buffer 0 is reused as T / lora_B scratch
  // in the epilogue (main loop is done with it by then).
  __shared__ __align__(16) unsigned short lds_x [2][128 * LDSS];
  __shared__ __align__(16) unsigned short lds_w [2][128 * LDSS];
  __shared__ __align__(16) unsigned short lds_la[2][ 16 * LDSS];

  const int tid  = threadIdx.x;
  const int lane = tid & 31;
  const int wid  = tid >> 5;       // 0..7
  const int wm   = wid >> 1;       // wave M row (0..3) -> offset wm*32
  const int wn   = wid & 1;        // wave N col (0..1) -> offset wn*64

  const int tileN = blockIdx.x * 128;
  const int tileM = blockIdx.y * 128;

  const float tabreg = NF4_TAB[lane & 15];   // lane-striped NF4 table

  v8f c[2][4];
  v8f tacc[2];
#pragma unroll
  for (int i = 0; i < 2; ++i) {
#pragma unroll
    for (int j = 0; j < 4; ++j) c[i][j] = {};
    tacc[i] = {};
  }

  // ---- Pipeline prologue: stage K-step 0 into buffer 0 -------------------
  TileRegs r;
  issue_loads(r, x, codes, scales, lora_A, tileM, tileN, 0, tid);
  store_tiles(r, lds_x[0], lds_w[0], lds_la[0], tabreg, tid);

  int cur = 0;
  for (int k0 = 0; k0 < KDIM; k0 += 32) {
    __syncthreads();   // buffer[cur] filled & buffer[cur^1] drained
    const bool more = (k0 + 32) < KDIM;

    // 1) Issue next tile's global loads (no wait yet).
    if (more)
      issue_loads(r, x, codes, scales, lora_A, tileM, tileN, k0 + 32, tid);

    // 2) Compute on current buffer — hides the global-load latency.
    const unsigned short* bx  = lds_x[cur];
    const unsigned short* bw  = lds_w[cur];
    const unsigned short* bla = lds_la[cur];

    v16bf a0 = ld_frag(bx, wm * 32 + 0,  lane);
    v16bf a1 = ld_frag(bx, wm * 32 + 16, lane);
    v16bf b0 = ld_frag(bw, wn * 64 + 0,  lane);
    v16bf b1 = ld_frag(bw, wn * 64 + 16, lane);
    v16bf b2 = ld_frag(bw, wn * 64 + 32, lane);
    v16bf b3 = ld_frag(bw, wn * 64 + 48, lane);

    c[0][0] = __builtin_amdgcn_wmma_f32_16x16x32_bf16(false, a0, false, b0, (short)0, c[0][0], false, false);
    c[0][1] = __builtin_amdgcn_wmma_f32_16x16x32_bf16(false, a0, false, b1, (short)0, c[0][1], false, false);
    c[0][2] = __builtin_amdgcn_wmma_f32_16x16x32_bf16(false, a0, false, b2, (short)0, c[0][2], false, false);
    c[0][3] = __builtin_amdgcn_wmma_f32_16x16x32_bf16(false, a0, false, b3, (short)0, c[0][3], false, false);
    c[1][0] = __builtin_amdgcn_wmma_f32_16x16x32_bf16(false, a1, false, b0, (short)0, c[1][0], false, false);
    c[1][1] = __builtin_amdgcn_wmma_f32_16x16x32_bf16(false, a1, false, b1, (short)0, c[1][1], false, false);
    c[1][2] = __builtin_amdgcn_wmma_f32_16x16x32_bf16(false, a1, false, b2, (short)0, c[1][2], false, false);
    c[1][3] = __builtin_amdgcn_wmma_f32_16x16x32_bf16(false, a1, false, b3, (short)0, c[1][3], false, false);

    if (wn == 0) {   // wave-uniform branch: EXEC stays all-ones inside
      v16bf la = ld_frag(bla, 0, lane);
      tacc[0] = __builtin_amdgcn_wmma_f32_16x16x32_bf16(false, a0, false, la, (short)0, tacc[0], false, false);
      tacc[1] = __builtin_amdgcn_wmma_f32_16x16x32_bf16(false, a1, false, la, (short)0, tacc[1], false, false);
    }

    // 3) Wait for the staged loads, convert, store into the other buffer.
    if (more)
      store_tiles(r, lds_x[cur ^ 1], lds_w[cur ^ 1], lds_la[cur ^ 1], tabreg, tid);
    cur ^= 1;
  }

  // ---- LoRA epilogue: out += (2*T) @ lora_B^T via WMMA -------------------
  unsigned short* ldsT = lds_x[0];   // reuse: T as bf16, K-padded with zeros
  unsigned short* ldsB = lds_w[0];   // reuse: lora_B tile, K-padded with zeros

  __syncthreads();                   // main-loop fragment reads finished
  {
    const U4 z = {0, 0, 0, 0};
    for (int i = tid; i < (128 * LDSS) / 8; i += 256) {   // b128 zero-fill
      *(U4*)&ldsT[i * 8] = z;
      *(U4*)&ldsB[i * 8] = z;
    }
  }
  __syncthreads();

  for (int i = tid; i < 128 * 16; i += 256) {
    int n = i >> 4, rr = i & 15;
    *(__bf16*)&ldsB[n * LDSS + rr] = (__bf16)lora_B[(size_t)(tileN + n) * 16 + rr];
  }
  if (wn == 0) {
#pragma unroll
    for (int sub = 0; sub < 2; ++sub) {
#pragma unroll
      for (int e = 0; e < 8; ++e) {
        // C-frag layout: M = e + 8*(lane>>4), N(r) = lane&15
        int m = wm * 32 + sub * 16 + e + ((lane >> 4) << 3);
        *(__bf16*)&ldsT[m * LDSS + (lane & 15)] = (__bf16)(2.0f * tacc[sub][e]);
      }
    }
  }
  __syncthreads();

#pragma unroll
  for (int i = 0; i < 2; ++i) {
    v16bf at = ld_frag(ldsT, wm * 32 + i * 16, lane);   // K=r (16..31 zero)
#pragma unroll
    for (int j = 0; j < 4; ++j) {
      v16bf bl = ld_frag(ldsB, wn * 64 + j * 16, lane);
      c[i][j] = __builtin_amdgcn_wmma_f32_16x16x32_bf16(false, at, false, bl, (short)0, c[i][j], false, false);
    }
  }

  // ---- Store: C-frag element e -> (M = e + 8*(lane>>4), N = lane&15) -----
#pragma unroll
  for (int i = 0; i < 2; ++i) {
#pragma unroll
    for (int j = 0; j < 4; ++j) {
      int gm0 = tileM + wm * 32 + i * 16 + ((lane >> 4) << 3);
      int gn0 = tileN + wn * 64 + j * 16 + (lane & 15);
#pragma unroll
      for (int e = 0; e < 8; ++e) {
        out[(size_t)(gm0 + e) * NDIM + gn0] = c[i][j][e];
      }
    }
  }
}

extern "C" void kernel_launch(void* const* d_in, const int* in_sizes, int n_in,
                              void* d_out, int out_size, void* d_ws, size_t ws_size,
                              hipStream_t stream) {
  const float* x      = (const float*)d_in[0];
  const int*   codes  = (const int*)  d_in[1];
  const float* scales = (const float*)d_in[2];
  const float* lora_A = (const float*)d_in[3];
  const float* lora_B = (const float*)d_in[4];
  float* out = (float*)d_out;

  const int M = in_sizes[0] / KDIM;          // B*S = 8192
  dim3 grid(NDIM / 128, M / 128);            // 32 x 64 workgroups
  qlora_wmma_kernel<<<grid, 256, 0, stream>>>(x, codes, scales, lora_A, lora_B, out);
}
